// Expert_65343632441649
// MI455X (gfx1250) — compile-verified, matrix operand order checked by reading, not run on hardware
//
#include <hip/hip_runtime.h>
#include <hip/hip_bf16.h>

// Problem dims (match reference)
#define BDIM  8192
#define CDIM  4096
#define GDIM  8
#define HIDD  1024
#define CTRLN 64
#define NDIM  (GDIM * HIDD)   // 8192 = concatenated expert hidden dim

typedef __bf16 bf16;
typedef __bf16 v16bf __attribute__((ext_vector_type(16)));
typedef __bf16 v8bf  __attribute__((ext_vector_type(8)));
typedef __bf16 v4bf  __attribute__((ext_vector_type(4)));
typedef float  v8f   __attribute__((ext_vector_type(8)));
typedef float  v4f   __attribute__((ext_vector_type(4)));
typedef unsigned int v4u __attribute__((ext_vector_type(4)));
typedef int    v8i   __attribute__((ext_vector_type(8)));
typedef int    v4i   __attribute__((ext_vector_type(4)));

#if defined(__gfx1250__) && __has_builtin(__builtin_amdgcn_tensor_load_to_lds) && \
    __has_builtin(__builtin_amdgcn_s_wait_tensorcnt)
#define USE_TDM 1
#else
#define USE_TDM 0
#endif

// ---------------------------------------------------------------------------
// WMMA fragment load from an LDS tile stored as [row][32] bf16 (row = M for A,
// row = N for pre-transposed B). Matches CDNA5 16-bit A/B operand layout:
//   lanes 0-15  : row = r0+lane,    elems = K[0..7]  ++ K[16..23]
//   lanes 16-31 : row = r0+lane-16, elems = K[8..15] ++ K[24..31]
// ---------------------------------------------------------------------------
static __device__ inline v16bf ld_frag(const bf16* __restrict__ tile, int r0, int lane) {
    const bf16* p = tile + (size_t)(r0 + (lane & 15)) * 32 + ((lane >> 4) * 8);
    v8bf a = *(const v8bf*)(p);        // K 0-7   (rel)
    v8bf b = *(const v8bf*)(p + 16);   // K 16-23 (rel)
    v16bf r;
#pragma unroll
    for (int i = 0; i < 8; ++i) { r[i] = a[i]; r[i + 8] = b[i]; }
    return r;
}

static __device__ inline v8f wmma_bf16(v16bf a, v16bf b, v8f c) {
    return __builtin_amdgcn_wmma_f32_16x16x32_bf16(false, a, false, b, (short)0, c,
                                                   false, false);
}

#if USE_TDM
// ---------------------------------------------------------------------------
// Tensor Data Mover: 2D bf16 tile load (tile_k x tile_rows) global -> LDS.
// D# packed per CDNA5 ISA 8.3/8.4: group0 = {flags, lds_addr, global_addr, type},
// group1 = {data_size, tensor dims, tile dims, strides}. 2D -> upper groups zero.
// Toolchain here exposes the 6-arg builtin: (v4u, v8i, v4i, v4i, v8i, i32 cpol).
// Note: for __shared__ pointers the low 32 bits of the generic address are the
// LDS byte offset (flat-aperture truncation rule), which is what D#.lds_addr wants.
// ---------------------------------------------------------------------------
static __device__ __attribute__((always_inline)) void tdm_load_tile_bf16(
    const bf16* gsrc, void* lds_dst, unsigned tile_k, unsigned tile_rows,
    unsigned tensor_k, unsigned tensor_rows, unsigned row_stride_elems)
{
    unsigned long long ga = (unsigned long long)(__SIZE_TYPE__)gsrc;
    unsigned lds_addr = (unsigned)(unsigned long long)(__SIZE_TYPE__)lds_dst;

    v4u g0;
    g0[0] = 1u;                                          // count=1 user descriptor
    g0[1] = lds_addr;                                    // lds_addr[31:0]
    g0[2] = (unsigned)(ga & 0xffffffffu);                // global_addr[31:0]
    g0[3] = (unsigned)((ga >> 32) & 0x01ffffffu)         // global_addr[56:32]
          | (2u << 30);                                  // type=2 ("image")

    v8i g1;
    g1[0] = (int)(1u << 16);                             // mask=0, data_size=1 (2B)
    g1[1] = (int)((tensor_k & 0xffffu) << 16);           // tensor_dim0[15:0]
    g1[2] = (int)(((tensor_k >> 16) & 0xffffu)           // tensor_dim0[31:16]
          | ((tensor_rows & 0xffffu) << 16));            // tensor_dim1[15:0]
    g1[3] = (int)(((tensor_rows >> 16) & 0xffffu)        // tensor_dim1[31:16]
          | ((tile_k & 0xffffu) << 16));                 // tile_dim0
    g1[4] = (int)(tile_rows & 0xffffu);                  // tile_dim1 (tile_dim2=0)
    g1[5] = (int)row_stride_elems;                       // tensor_dim0_stride[31:0]
    g1[6] = 0;                                           // stride hi + dim1_stride lo
    g1[7] = 0;

    v4i z4; z4[0] = z4[1] = z4[2] = z4[3] = 0;
    v8i z8;
#pragma unroll
    for (int i = 0; i < 8; ++i) z8[i] = 0;
    __builtin_amdgcn_tensor_load_to_lds(g0, g1, z4, z4, z8, 0);
}
#endif

// ---------------------------------------------------------------------------
// Gate: choose = softmax(relu(y@wc1 + bc1) @ wc2 + bc2), one thread per row.
// ---------------------------------------------------------------------------
__global__ __launch_bounds__(256) void gate_softmax_kernel(
    const float* __restrict__ y, const float* __restrict__ wc1,
    const float* __restrict__ bc1, const float* __restrict__ wc2,
    const float* __restrict__ bc2, float* __restrict__ choose)
{
    __shared__ float swc1[CTRLN * GDIM];   // 512
    __shared__ float swc2[GDIM * GDIM];    // 64
    __shared__ float sbc1[GDIM];
    __shared__ float sbc2[GDIM];

    const int tid = threadIdx.x;
    swc1[tid]       = wc1[tid];
    swc1[tid + 256] = wc1[tid + 256];
    if (tid < GDIM * GDIM) swc2[tid] = wc2[tid];
    if (tid < GDIM) { sbc1[tid] = bc1[tid]; sbc2[tid] = bc2[tid]; }
    __syncthreads();

    const int b = blockIdx.x * 256 + tid;
    const float* yr = y + (size_t)b * CTRLN;

    float t[GDIM];
#pragma unroll
    for (int g = 0; g < GDIM; ++g) t[g] = sbc1[g];
    for (int c = 0; c < CTRLN; c += 4) {
        v4f yv = *(const v4f*)(yr + c);
#pragma unroll
        for (int j = 0; j < 4; ++j) {
            float yy = yv[j];
#pragma unroll
            for (int g = 0; g < GDIM; ++g) t[g] += yy * swc1[(c + j) * GDIM + g];
        }
    }
#pragma unroll
    for (int g = 0; g < GDIM; ++g) t[g] = fmaxf(t[g], 0.f);

    float s[GDIM];
#pragma unroll
    for (int g = 0; g < GDIM; ++g) s[g] = sbc2[g];
#pragma unroll
    for (int gp = 0; gp < GDIM; ++gp) {
#pragma unroll
        for (int g = 0; g < GDIM; ++g) s[g] += t[gp] * swc2[gp * GDIM + g];
    }
    float mx = s[0];
#pragma unroll
    for (int g = 1; g < GDIM; ++g) mx = fmaxf(mx, s[g]);
    float e[GDIM], sum = 0.f;
#pragma unroll
    for (int g = 0; g < GDIM; ++g) { e[g] = __expf(s[g] - mx); sum += e[g]; }
    const float inv = 1.f / sum;
#pragma unroll
    for (int g = 0; g < GDIM; ++g) choose[(size_t)b * GDIM + g] = e[g] * inv;
}

// ---------------------------------------------------------------------------
// Stage 1: Hs[b, g*1024+h] = relu(X @ W1[g] + b1[g]) * choose[b,g]   (bf16 out)
// 128x128 WG tile, K-step 32, double-buffered LDS, 8 waves -> 32x64 per wave.
// ---------------------------------------------------------------------------
__global__ __launch_bounds__(256) void expert_stage1_kernel(
    const float* __restrict__ X, const float* __restrict__ W1,
    const float* __restrict__ b1, const float* __restrict__ choose,
    bf16* __restrict__ Hs)
{
    __shared__ __align__(16) bf16 sA[2][128 * 32];
    __shared__ __align__(16) bf16 sB[2][128 * 32];   // stored transposed: [n][k]
    __shared__ float sScale[128];
    __shared__ float sBias[128];

    const int tid  = threadIdx.x;
    const int lane = tid & 31;
    const int wave = tid >> 5;
    const int wm   = wave & 3;    // 4 waves along M
    const int wn   = wave >> 2;   // 2 waves along N
    const int n0   = blockIdx.x * 128;
    const int m0   = blockIdx.y * 128;
    const int g    = n0 >> 10;          // expert id for this N tile
    const int h0   = n0 & (HIDD - 1);   // column offset inside expert

    const float* Ag = X + (size_t)m0 * CDIM;
    const float* Bm = W1 + (size_t)g * CDIM * HIDD + h0;   // [K=4096][HID], col h0

    if (tid < 128) {
        sScale[tid] = choose[(size_t)(m0 + tid) * GDIM + g];
        sBias[tid]  = b1[(size_t)g * HIDD + h0 + tid];
    }

    // staging coordinates (256 threads)
    const int ar = tid >> 1;            // A row 0..127
    const int ac = (tid & 1) * 16;      // A col group {0,16}
    const int bn = (tid & 31) * 4;      // B: 4 columns (n)
    const int bk = (tid >> 5) * 4;      // B: 4 rows (k)

    v8f zero;
#pragma unroll
    for (int i = 0; i < 8; ++i) zero[i] = 0.f;
    v8f acc[2][4];
#pragma unroll
    for (int mi = 0; mi < 2; ++mi)
#pragma unroll
        for (int ni = 0; ni < 4; ++ni) acc[mi][ni] = zero;

    float ra[16], rb[4][4];

    // ---- prologue: fetch k-tile 0 into regs, commit to buffer 0
#pragma unroll
    for (int i = 0; i < 4; ++i) {
        v4f v = *(const v4f*)(Ag + (size_t)ar * CDIM + ac + i * 4);
        ra[4 * i + 0] = v[0]; ra[4 * i + 1] = v[1]; ra[4 * i + 2] = v[2]; ra[4 * i + 3] = v[3];
    }
#pragma unroll
    for (int p = 0; p < 4; ++p) {
        v4f v = *(const v4f*)(Bm + (size_t)(bk + p) * HIDD + bn);
        rb[p][0] = v[0]; rb[p][1] = v[1]; rb[p][2] = v[2]; rb[p][3] = v[3];
    }
    {
        v8bf lo, hi;
#pragma unroll
        for (int i = 0; i < 8; ++i) { lo[i] = (bf16)ra[i]; hi[i] = (bf16)ra[8 + i]; }
        *(v8bf*)(&sA[0][ar * 32 + ac])     = lo;
        *(v8bf*)(&sA[0][ar * 32 + ac + 8]) = hi;
#pragma unroll
        for (int j = 0; j < 4; ++j) {   // transpose 4x4 block: contiguous-k b64 stores
            v4bf q;
#pragma unroll
            for (int p = 0; p < 4; ++p) q[p] = (bf16)rb[p][j];
            *(v4bf*)(&sB[0][(bn + j) * 32 + bk]) = q;
        }
    }
    __syncthreads();

    const int NK = CDIM / 32;
    for (int kt = 0; kt < NK; ++kt) {
        const int  cur  = kt & 1;
        const int  nxt  = cur ^ 1;
        const bool more = (kt + 1 < NK);
        if (more) {
            const int k0 = (kt + 1) * 32;
#pragma unroll
            for (int i = 0; i < 4; ++i) {
                v4f v = *(const v4f*)(Ag + (size_t)ar * CDIM + k0 + ac + i * 4);
                ra[4 * i + 0] = v[0]; ra[4 * i + 1] = v[1]; ra[4 * i + 2] = v[2]; ra[4 * i + 3] = v[3];
            }
#pragma unroll
            for (int p = 0; p < 4; ++p) {
                v4f v = *(const v4f*)(Bm + (size_t)(k0 + bk + p) * HIDD + bn);
                rb[p][0] = v[0]; rb[p][1] = v[1]; rb[p][2] = v[2]; rb[p][3] = v[3];
            }
            if (kt + 2 < NK) {
                __builtin_prefetch(Ag + (size_t)ar * CDIM + (kt + 2) * 32 + ac, 0, 0);
                __builtin_prefetch(Bm + (size_t)((kt + 2) * 32 + bk) * HIDD + bn, 0, 0);
            }
        }
        // ---- compute on current buffer (overlaps the global loads above)
        v16bf af[2], bfr[4];
#pragma unroll
        for (int mi = 0; mi < 2; ++mi) af[mi] = ld_frag(&sA[cur][0], wm * 32 + mi * 16, lane);
#pragma unroll
        for (int ni = 0; ni < 4; ++ni) bfr[ni] = ld_frag(&sB[cur][0], wn * 64 + ni * 16, lane);
#pragma unroll
        for (int mi = 0; mi < 2; ++mi)
#pragma unroll
            for (int ni = 0; ni < 4; ++ni)
                acc[mi][ni] = wmma_bf16(af[mi], bfr[ni], acc[mi][ni]);

        if (more) {
            v8bf lo, hi;
#pragma unroll
            for (int i = 0; i < 8; ++i) { lo[i] = (bf16)ra[i]; hi[i] = (bf16)ra[8 + i]; }
            *(v8bf*)(&sA[nxt][ar * 32 + ac])     = lo;
            *(v8bf*)(&sA[nxt][ar * 32 + ac + 8]) = hi;
#pragma unroll
            for (int j = 0; j < 4; ++j) {
                v4bf q;
#pragma unroll
                for (int p = 0; p < 4; ++p) q[p] = (bf16)rb[p][j];
                *(v4bf*)(&sB[nxt][(bn + j) * 32 + bk]) = q;
            }
        }
        __syncthreads();
    }

    // ---- epilogue: bias + relu + gate scale, store bf16
#pragma unroll
    for (int mi = 0; mi < 2; ++mi) {
#pragma unroll
        for (int ni = 0; ni < 4; ++ni) {
            const int lcol  = wn * 64 + ni * 16 + (lane & 15);
            const int rbase = wm * 32 + mi * 16 + ((lane >> 4) * 8);
#pragma unroll
            for (int r = 0; r < 8; ++r) {
                const int lrow = rbase + r;
                float v = acc[mi][ni][r] + sBias[lcol];
                v = fmaxf(v, 0.f) * sScale[lrow];
                Hs[(size_t)(m0 + lrow) * NDIM + (n0 + lcol)] = (bf16)v;
            }
        }
    }
}

// ---------------------------------------------------------------------------
// Stage 2: Out = Hs[B,8192](bf16) @ W2flat[8192,4096] + sum_g choose[b,g]*b2[g,:]
// A tiles (pure bf16 rectangles) are DMA'd by the Tensor Data Mover; wave 0
// issues tensor_load_to_lds for the next buffer while all waves run WMMA.
// ---------------------------------------------------------------------------
__global__ __launch_bounds__(256) void expert_stage2_kernel(
    const bf16* __restrict__ Hs, const float* __restrict__ W2,
    const float* __restrict__ b2, const float* __restrict__ choose,
    float* __restrict__ Out)
{
    __shared__ __align__(16) bf16 sA[2][128 * 32];
    __shared__ __align__(16) bf16 sB[2][128 * 32];   // transposed [n][k]
    __shared__ float sB2[GDIM * 128];
    __shared__ float sCh[128 * GDIM];

    const int tid  = threadIdx.x;
    const int lane = tid & 31;
    const int wave = tid >> 5;
    const int wm   = wave & 3;
    const int wn   = wave >> 2;
    const int n0   = blockIdx.x * 128;
    const int m0   = blockIdx.y * 128;

    const bf16*  Ag = Hs + (size_t)m0 * NDIM;
    const float* Bm = W2 + n0;                 // [K=8192][C=4096]

    if (tid < 128) {
#pragma unroll
        for (int g = 0; g < GDIM; ++g)
            sCh[tid * GDIM + g] = choose[(size_t)(m0 + tid) * GDIM + g];
    }
    {   // b2 tile: 8 x 128 floats, 256 threads x 4
        const int gg = tid >> 5;
        const int nn = (tid & 31) * 4;
        v4f v = *(const v4f*)(b2 + (size_t)gg * CDIM + n0 + nn);
        sB2[gg * 128 + nn + 0] = v[0]; sB2[gg * 128 + nn + 1] = v[1];
        sB2[gg * 128 + nn + 2] = v[2]; sB2[gg * 128 + nn + 3] = v[3];
    }

    const int ar = tid >> 1;
    const int ac = (tid & 1) * 16;
    const int bn = (tid & 31) * 4;
    const int bk = (tid >> 5) * 4;

    v8f zero;
#pragma unroll
    for (int i = 0; i < 8; ++i) zero[i] = 0.f;
    v8f acc[2][4];
#pragma unroll
    for (int mi = 0; mi < 2; ++mi)
#pragma unroll
        for (int ni = 0; ni < 4; ++ni) acc[mi][ni] = zero;

    float rb[4][4];
#if !USE_TDM
    uint4 ua0, ua1;
#endif

    // ---- prologue: tile 0
#if USE_TDM
    if (wave == 0)
        tdm_load_tile_bf16(Ag, &sA[0][0], 32u, 128u, (unsigned)NDIM, (unsigned)BDIM,
                           (unsigned)NDIM);
#else
    {
        const uint4* pa = (const uint4*)(Ag + (size_t)ar * NDIM + ac);
        ua0 = pa[0]; ua1 = pa[1];
        *(uint4*)(&sA[0][ar * 32 + ac])     = ua0;
        *(uint4*)(&sA[0][ar * 32 + ac + 8]) = ua1;
    }
#endif
#pragma unroll
    for (int p = 0; p < 4; ++p) {
        v4f v = *(const v4f*)(Bm + (size_t)(bk + p) * CDIM + bn);
        rb[p][0] = v[0]; rb[p][1] = v[1]; rb[p][2] = v[2]; rb[p][3] = v[3];
    }
#pragma unroll
    for (int j = 0; j < 4; ++j) {
        v4bf q;
#pragma unroll
        for (int p = 0; p < 4; ++p) q[p] = (bf16)rb[p][j];
        *(v4bf*)(&sB[0][(bn + j) * 32 + bk]) = q;
    }
#if USE_TDM
    if (wave == 0) __builtin_amdgcn_s_wait_tensorcnt((short)0);
#endif
    __syncthreads();

    const int NK = NDIM / 32;
    for (int kt = 0; kt < NK; ++kt) {
        const int  cur  = kt & 1;
        const int  nxt  = cur ^ 1;
        const bool more = (kt + 1 < NK);
        if (more) {
            const int k0 = (kt + 1) * 32;
#if USE_TDM
            if (wave == 0)
                tdm_load_tile_bf16(Ag + k0, &sA[nxt][0], 32u, 128u, (unsigned)NDIM,
                                   (unsigned)BDIM, (unsigned)NDIM);
#else
            const uint4* pa = (const uint4*)(Ag + (size_t)ar * NDIM + k0 + ac);
            ua0 = pa[0]; ua1 = pa[1];
#endif
#pragma unroll
            for (int p = 0; p < 4; ++p) {
                v4f v = *(const v4f*)(Bm + (size_t)(k0 + bk + p) * CDIM + bn);
                rb[p][0] = v[0]; rb[p][1] = v[1]; rb[p][2] = v[2]; rb[p][3] = v[3];
            }
            if (kt + 2 < NK)
                __builtin_prefetch(Bm + (size_t)((kt + 2) * 32 + bk) * CDIM + bn, 0, 0);
        }
        // ---- compute on current buffer
        v16bf af[2], bfr[4];
#pragma unroll
        for (int mi = 0; mi < 2; ++mi) af[mi] = ld_frag(&sA[cur][0], wm * 32 + mi * 16, lane);
#pragma unroll
        for (int ni = 0; ni < 4; ++ni) bfr[ni] = ld_frag(&sB[cur][0], wn * 64 + ni * 16, lane);
#pragma unroll
        for (int mi = 0; mi < 2; ++mi)
#pragma unroll
            for (int ni = 0; ni < 4; ++ni)
                acc[mi][ni] = wmma_bf16(af[mi], bfr[ni], acc[mi][ni]);

        if (more) {
#if !USE_TDM
            *(uint4*)(&sA[nxt][ar * 32 + ac])     = ua0;
            *(uint4*)(&sA[nxt][ar * 32 + ac + 8]) = ua1;
#endif
#pragma unroll
            for (int j = 0; j < 4; ++j) {
                v4bf q;
#pragma unroll
                for (int p = 0; p < 4; ++p) q[p] = (bf16)rb[p][j];
                *(v4bf*)(&sB[nxt][(bn + j) * 32 + bk]) = q;
            }
        }
#if USE_TDM
        if (more && wave == 0) __builtin_amdgcn_s_wait_tensorcnt((short)0);
#endif
        __syncthreads();
    }

    // ---- epilogue: add folded gate-weighted bias, store f32
#pragma unroll
    for (int mi = 0; mi < 2; ++mi) {
#pragma unroll
        for (int ni = 0; ni < 4; ++ni) {
            const int lcol  = wn * 64 + ni * 16 + (lane & 15);
            const int rbase = wm * 32 + mi * 16 + ((lane >> 4) * 8);
#pragma unroll
            for (int r = 0; r < 8; ++r) {
                const int lrow = rbase + r;
                float v = acc[mi][ni][r];
#pragma unroll
                for (int g = 0; g < GDIM; ++g)
                    v += sCh[lrow * GDIM + g] * sB2[g * 128 + lcol];
                Out[(size_t)(m0 + lrow) * CDIM + (n0 + lcol)] = v;
            }
        }
    }
}

// ---------------------------------------------------------------------------
extern "C" void kernel_launch(void* const* d_in, const int* in_sizes, int n_in,
                              void* d_out, int out_size, void* d_ws, size_t ws_size,
                              hipStream_t stream)
{
    (void)in_sizes; (void)n_in; (void)out_size; (void)ws_size;
    const float* x   = (const float*)d_in[0];
    const float* y   = (const float*)d_in[1];
    const float* w1  = (const float*)d_in[2];
    const float* b1  = (const float*)d_in[3];
    const float* w2  = (const float*)d_in[4];
    const float* b2  = (const float*)d_in[5];
    const float* wc1 = (const float*)d_in[6];
    const float* bc1 = (const float*)d_in[7];
    const float* wc2 = (const float*)d_in[8];
    const float* bc2 = (const float*)d_in[9];
    float* out = (float*)d_out;

    float* choose = (float*)d_ws;                               // 8192*8 f32 = 256 KB
    bf16*  Hs     = (bf16*)((char*)d_ws + (512u << 10));        // 8192*8192 bf16 = 128 MB

    gate_softmax_kernel<<<BDIM / 256, 256, 0, stream>>>(y, wc1, bc1, wc2, bc2, choose);
    expert_stage1_kernel<<<dim3(NDIM / 128, BDIM / 128), 256, 0, stream>>>(x, w1, b1, choose, Hs);
    expert_stage2_kernel<<<dim3(CDIM / 128, BDIM / 128), 256, 0, stream>>>(Hs, w2, b2, choose, out);
}
